// SelfAttention_549755814308
// MI455X (gfx1250) — matrix operand order repaired and checked by reading, not compiled, not run on hardware
//
#include <hip/hip_runtime.h>
#include <stdint.h>

// ---------------------------------------------------------------------------
// Types for CDNA5 WMMA (wave32)
// ---------------------------------------------------------------------------
typedef __attribute__((ext_vector_type(16))) __bf16        v16bf;
typedef __attribute__((ext_vector_type(8)))  float         v8f;
typedef __attribute__((ext_vector_type(4)))  unsigned int  u32x4;
typedef __attribute__((ext_vector_type(4)))  int           i32x4v;

struct Frag32B { u32x4 lo, hi; };
static_assert(sizeof(Frag32B) == sizeof(v16bf), "frag size mismatch");

#if defined(__gfx1250__) && \
    __has_builtin(__builtin_amdgcn_global_load_async_to_lds_b128) && \
    __has_builtin(__builtin_amdgcn_s_wait_asynccnt)
#define HAS_ASYNC_LDS 1
typedef __attribute__((address_space(1))) i32x4v g_i32x4;   // global int4
typedef __attribute__((address_space(3))) i32x4v l_i32x4;   // LDS int4
#endif

// 16-byte cooperative copy global -> LDS (async path when available)
__device__ __forceinline__ void stage16(const void* g, void* l) {
#ifdef HAS_ASYNC_LDS
  __builtin_amdgcn_global_load_async_to_lds_b128(
      (g_i32x4*)(uintptr_t)g, (l_i32x4*)(unsigned)(uintptr_t)l, 0, 0);
#else
  *(u32x4*)l = *(const u32x4*)g;
#endif
}
__device__ __forceinline__ void stage_fence() {
#ifdef HAS_ASYNC_LDS
  __builtin_amdgcn_s_wait_asynccnt(0);
#endif
}

__device__ __forceinline__ unsigned short f2bf(float f) {
  unsigned int u = __builtin_bit_cast(unsigned int, f);
  u += 0x7FFFu + ((u >> 16) & 1u);          // round-to-nearest-even
  return (unsigned short)(u >> 16);
}

__device__ __forceinline__ v8f zero8() {
  v8f v;
#pragma unroll
  for (int i = 0; i < 8; ++i) v[i] = 0.0f;
  return v;
}

// A-matrix fragment (16x32 bf16, ISA 7.12.2): lane owns row M=lane&15;
// lanes 0-15 hold K=[0..7]+[16..23], lanes 16-31 hold K=[8..15]+[24..31].
__device__ __forceinline__ v16bf load_fragA(const unsigned short* p, int khalf) {
  Frag32B t;
  t.lo = *(const u32x4*)(p + khalf * 8);
  t.hi = *(const u32x4*)(p + 16 + khalf * 8);
  return __builtin_bit_cast(v16bf, t);
}

// B-matrix fragment (32x16 bf16): lane owns column N=lane&15; lanes 0-15 hold
// K=[0..7]+[16..23], lanes 16-31 hold K=[8..15]+[24..31].  p = (BT row base) + k0,
// i.e. two 16-byte runs at p + khalf*8 and p + 16 + khalf*8.
__device__ __forceinline__ v16bf load_fragB(const unsigned short* p, int khalf) {
  Frag32B t;
  t.lo = *(const u32x4*)(p + khalf * 8);
  t.hi = *(const u32x4*)(p + 16 + khalf * 8);
  return __builtin_bit_cast(v16bf, t);
}

__device__ __forceinline__ v8f wmma_bf16(v16bf a, v16bf b, v8f c) {
  // (neg_a, A, neg_b, B, c_mod, C, reuse_a, reuse_b)
  return __builtin_amdgcn_wmma_f32_16x16x32_bf16(false, a, false, b,
                                                 (short)0, c, false, false);
}

// ---------------------------------------------------------------------------
// Kernel 0a: fp32 -> bf16 elementwise (weights)
// ---------------------------------------------------------------------------
__global__ void k_f32_to_bf16(const float* __restrict__ src,
                              unsigned short* __restrict__ dst, int n) {
  int i = blockIdx.x * blockDim.x + threadIdx.x;
  if (i < n) dst[i] = f2bf(src[i]);
}

// ---------------------------------------------------------------------------
// Kernel 0b: x [B,C,N] fp32 -> xT [B,N,C] bf16 (32x32 LDS tile transpose)
// ---------------------------------------------------------------------------
__global__ void k_transpose_bf16(const float* __restrict__ x,
                                 unsigned short* __restrict__ xT) {
  __shared__ float tile[32][33];
  const int b  = blockIdx.z;
  const int n0 = blockIdx.x * 32;
  const int c0 = blockIdx.y * 32;
#pragma unroll
  for (int r = 0; r < 4; ++r) {
    int c = c0 + threadIdx.y + r * 8;
    tile[threadIdx.y + r * 8][threadIdx.x] =
        x[((size_t)b * 256 + c) * 4096 + n0 + threadIdx.x];
  }
  __syncthreads();
#pragma unroll
  for (int r = 0; r < 4; ++r) {
    int n = n0 + threadIdx.y + r * 8;
    xT[((size_t)b * 4096 + n) * 256 + c0 + threadIdx.x] =
        f2bf(tile[threadIdx.x][threadIdx.y + r * 8]);
  }
}

// ---------------------------------------------------------------------------
// Kernel 1: generic bf16 GEMM  Out[m,n] = sum_k A[m,k]*BT[n,k] + bias
//   A  : row-major M x 256, BT : row-major N x 256 (i.e. B transposed)
//   Block = 256 threads = 8 waves; block tile 128(M) x 64(N);
//   wave tile 16(M) x 64(N) = 4 WMMA accumulators; K loop 8 x 32.
// ---------------------------------------------------------------------------
__global__ __launch_bounds__(256) void k_gemm_bf16(
    const unsigned short* __restrict__ A,  long long strideA,
    const unsigned short* __restrict__ BT, long long strideBT,
    const float* __restrict__ bias, int biasPerRow,
    unsigned short* __restrict__ Out, long long strideOut, int ldo) {
  const int tid  = threadIdx.x;
  const int wid  = tid >> 5;
  const int lane = tid & 31;
  const int lr    = lane & 15;
  const int khalf = lane >> 4;

  const unsigned short* Ab = A  + (size_t)blockIdx.z * strideA;
  const unsigned short* Bb = BT + (size_t)blockIdx.z * strideBT;
  unsigned short*       Ob = Out + (size_t)blockIdx.z * strideOut;

  const int mRow  = blockIdx.y * 128 + wid * 16 + lr;  // A row this lane feeds
  const int nBase = blockIdx.x * 64;

  v8f acc[4];
#pragma unroll
  for (int t = 0; t < 4; ++t) acc[t] = zero8();

#pragma unroll
  for (int k0 = 0; k0 < 256; k0 += 32) {
    v16bf a = load_fragA(Ab + (size_t)mRow * 256 + k0, khalf);
#pragma unroll
    for (int t = 0; t < 4; ++t) {
      v16bf bf = load_fragB(Bb + (size_t)(nBase + t * 16 + lr) * 256 + k0, khalf);
      acc[t] = wmma_bf16(a, bf, acc[t]);
    }
  }

  // C/D layout: lane holds col n = lr, rows m = khalf*8 + r
  const int mOut = blockIdx.y * 128 + wid * 16 + khalf * 8;
#pragma unroll
  for (int t = 0; t < 4; ++t) {
    int n = nBase + t * 16 + lr;
#pragma unroll
    for (int r = 0; r < 8; ++r) {
      int m = mOut + r;
      float v = acc[t][r] + (biasPerRow ? bias[m] : bias[n]);
      Ob[(size_t)m * ldo + n] = f2bf(v);
    }
  }
}

// ---------------------------------------------------------------------------
// Kernel 2: flash attention (B=4, C=256, N=4096)
//   QT,KT : [B,N,C] bf16 ; V : [B,C,N] bf16 ; out fp32 [B,C,N]
//   Block = 256 thr (8 waves), each wave owns 16 query rows, block 128 rows.
//   j loop in blocks of 32: S (2 WMMA tiles) -> online softmax -> PV (16 tiles)
// ---------------------------------------------------------------------------
__global__ __launch_bounds__(256) void k_flash_attn(
    const unsigned short* __restrict__ QT,
    const unsigned short* __restrict__ KT,
    const unsigned short* __restrict__ V,
    const float* __restrict__ x,
    const float* __restrict__ gamma,
    float* __restrict__ out) {
  __shared__ __align__(16) unsigned short Kt[32 * 256];   // K rows j0..j0+31
  __shared__ __align__(16) unsigned short Vt[256 * 32];   // V[:, j0..j0+31]
  __shared__ __align__(16) unsigned short Pt[8][16 * 32]; // per-wave P patch

  const int tid   = threadIdx.x;
  const int wid   = tid >> 5;
  const int lane  = tid & 31;
  const int lr    = lane & 15;
  const int khalf = lane >> 4;
  const int b     = blockIdx.y;
  const int iBase = blockIdx.x * 128 + wid * 16;

  const unsigned short* Qb = QT + (size_t)b * 4096 * 256;
  const unsigned short* Kb = KT + (size_t)b * 4096 * 256;
  const unsigned short* Vb = V  + (size_t)b * 256 * 4096;

  // Preload the 16x256 Q block for this wave as 8 A-fragments.
  v16bf qf[8];
#pragma unroll
  for (int kk = 0; kk < 8; ++kk)
    qf[kk] = load_fragA(Qb + (size_t)(iBase + lr) * 256 + kk * 32, khalf);

  v8f o[16];
#pragma unroll
  for (int t = 0; t < 16; ++t) o[t] = zero8();
  float mrow[8], lsum[8];
#pragma unroll
  for (int r = 0; r < 8; ++r) { mrow[r] = -3.0e38f; lsum[r] = 0.0f; }

  for (int j0 = 0; j0 < 4096; j0 += 32) {
    // ---- cooperative async staging: 32x256 K rows, 256x32 V slice ----
#pragma unroll
    for (int q = 0; q < 4; ++q) {
      int chunk = tid + q * 256;               // 1024 chunks of 16B
      int row = chunk >> 5, off = (chunk & 31) * 8;
      stage16(Kb + (size_t)(j0 + row) * 256 + off, &Kt[row * 256 + off]);
    }
#pragma unroll
    for (int q = 0; q < 4; ++q) {
      int chunk = tid + q * 256;
      int c = chunk >> 2, off = (chunk & 3) * 8;
      stage16(Vb + (size_t)c * 4096 + j0 + off, &Vt[c * 32 + off]);
    }
    stage_fence();        // s_wait_asynccnt 0 (this wave's async ops done)
    __syncthreads();      // whole block's staging visible

    // ---- S = Q . K^T for 16 rows x 32 cols ----
    v8f s0 = zero8(), s1 = zero8();
#pragma unroll
    for (int kk = 0; kk < 8; ++kk) {
      v16bf b0 = load_fragB(&Kt[(lr)      * 256 + kk * 32], khalf);
      v16bf b1 = load_fragB(&Kt[(16 + lr) * 256 + kk * 32], khalf);
      s0 = wmma_bf16(qf[kk], b0, s0);
      s1 = wmma_bf16(qf[kk], b1, s1);
    }

    // ---- online softmax: row = khalf*8 + r, cols spread over 16 lanes ----
    float sc[8];
#pragma unroll
    for (int r = 0; r < 8; ++r) {
      float a = fmaxf(s0[r], s1[r]);
      a = fmaxf(a, __shfl_xor(a, 1, 16));
      a = fmaxf(a, __shfl_xor(a, 2, 16));
      a = fmaxf(a, __shfl_xor(a, 4, 16));
      a = fmaxf(a, __shfl_xor(a, 8, 16));
      float nm = fmaxf(mrow[r], a);
      sc[r]  = __expf(mrow[r] - nm);
      float p0 = __expf(s0[r] - nm);
      float p1 = __expf(s1[r] - nm);
      s0[r] = p0; s1[r] = p1;
      float rs = p0 + p1;
      rs += __shfl_xor(rs, 1, 16);
      rs += __shfl_xor(rs, 2, 16);
      rs += __shfl_xor(rs, 4, 16);
      rs += __shfl_xor(rs, 8, 16);
      lsum[r] = lsum[r] * sc[r] + rs;
      mrow[r] = nm;
    }
#pragma unroll
    for (int t = 0; t < 16; ++t) {
#pragma unroll
      for (int r = 0; r < 8; ++r) o[t][r] *= sc[r];
    }

    // ---- C/D layout -> A-frag layout for P via per-wave LDS patch ----
    unsigned short* pw = &Pt[wid][0];
#pragma unroll
    for (int r = 0; r < 8; ++r) {
      int row = khalf * 8 + r;
      pw[row * 32 + lr]      = f2bf(s0[r]);
      pw[row * 32 + 16 + lr] = f2bf(s1[r]);
    }
    // same-wave DS ops are in-order; no barrier needed for this patch
    v16bf pf = load_fragA(pw + lr * 32, khalf);

    // ---- O += P . V^T over all 16 channel tiles ----
#pragma unroll
    for (int t = 0; t < 16; ++t) {
      v16bf vf = load_fragB(&Vt[(t * 16 + lr) * 32], khalf);
      o[t] = wmma_bf16(pf, vf, o[t]);
    }
    __syncthreads();   // protect Kt/Vt before next stage
  }

  // ---- epilogue: normalize, gamma * out + x, store fp32 ----
  const float g = gamma[0];
  float inv[8];
#pragma unroll
  for (int r = 0; r < 8; ++r) inv[r] = 1.0f / lsum[r];
#pragma unroll
  for (int t = 0; t < 16; ++t) {
    int c = t * 16 + lr;
#pragma unroll
    for (int r = 0; r < 8; ++r) {
      int i = iBase + khalf * 8 + r;
      size_t idx = ((size_t)b * 256 + c) * 4096 + i;
      out[idx] = g * (o[t][r] * inv[r]) + x[idx];
    }
  }
}

// ---------------------------------------------------------------------------
// Host launcher.  Inputs: x,Wq,bq,Wk,bk,Wv,bv,gamma (all fp32).
// Workspace layout (bf16 ushorts): xT | QT | KT | V | Wq | Wk | Wv  (~34 MB)
// ---------------------------------------------------------------------------
extern "C" void kernel_launch(void* const* d_in, const int* in_sizes, int n_in,
                              void* d_out, int out_size, void* d_ws, size_t ws_size,
                              hipStream_t stream) {
  const float* x     = (const float*)d_in[0];
  const float* Wq    = (const float*)d_in[1];
  const float* bq    = (const float*)d_in[2];
  const float* Wk    = (const float*)d_in[3];
  const float* bk    = (const float*)d_in[4];
  const float* Wv    = (const float*)d_in[5];
  const float* bv    = (const float*)d_in[6];
  const float* gamma = (const float*)d_in[7];
  float* out = (float*)d_out;

  unsigned short* ws = (unsigned short*)d_ws;
  const size_t S = (size_t)4 * 4096 * 256;      // per-tensor bf16 elements
  unsigned short* xT  = ws;
  unsigned short* QT  = ws + S;
  unsigned short* KT  = ws + 2 * S;
  unsigned short* Vb  = ws + 3 * S;
  unsigned short* Wqb = ws + 4 * S;
  unsigned short* Wkb = Wqb + 65536;
  unsigned short* Wvb = Wkb + 65536;

  // weights -> bf16
  k_f32_to_bf16<<<256, 256, 0, stream>>>(Wq, Wqb, 65536);
  k_f32_to_bf16<<<256, 256, 0, stream>>>(Wk, Wkb, 65536);
  k_f32_to_bf16<<<256, 256, 0, stream>>>(Wv, Wvb, 65536);

  // x -> xT bf16 [B,N,C]
  k_transpose_bf16<<<dim3(128, 8, 4), dim3(32, 8), 0, stream>>>(x, xT);

  const long long sX = (long long)4096 * 256;
  // QT[n,o] = xT . Wq^T + bq   (bias over GEMM-N = channel)
  k_gemm_bf16<<<dim3(4, 32, 4), 256, 0, stream>>>(xT, sX, Wqb, 0, bq, 0, QT, sX, 256);
  k_gemm_bf16<<<dim3(4, 32, 4), 256, 0, stream>>>(xT, sX, Wkb, 0, bk, 0, KT, sX, 256);
  // V[o,n]  = Wv . x + bv      (bias over GEMM-M = channel)
  k_gemm_bf16<<<dim3(64, 2, 4), 256, 0, stream>>>(Wvb, 0, xT, sX, bv, 1, Vb, sX, 4096);

  // fused attention + gamma*out + x
  k_flash_attn<<<dim3(32, 4), 256, 0, stream>>>(QT, KT, Vb, x, gamma, out);
}
